// MTMD_Domain_Expert_Add_6227702579383
// MI455X (gfx1250) — matrix operand order repaired and checked by reading, not compiled, not run on hardware
//
#include <hip/hip_runtime.h>

#define BN 65536
#define EPSF 1e-5f

typedef float f32x8 __attribute__((ext_vector_type(8)));
typedef __bf16 bf16x16 __attribute__((ext_vector_type(16)));
typedef unsigned int u32x8 __attribute__((ext_vector_type(8)));

__device__ __forceinline__ unsigned short f2bf(float f) {
  unsigned int u = __builtin_bit_cast(unsigned int, f);
  u += 0x7FFFu + ((u >> 16) & 1u);
  return (unsigned short)(u >> 16);
}
__device__ __forceinline__ float bf2f(unsigned short h) {
  unsigned int u = ((unsigned int)h) << 16;
  return __builtin_bit_cast(float, u);
}

__device__ __forceinline__ f32x8 wmma_bf16(bf16x16 a, bf16x16 b, f32x8 c) {
  return __builtin_amdgcn_wmma_f32_16x16x32_bf16(false, a, false, b, (short)0, c,
                                                 false, false);
}

// Async 16B global->LDS copy (GLOBAL_LOAD_ASYNC_TO_LDS_B128, ASYNCcnt).
// LDS destination = low 32 bits of the generic pointer (ISA 10.2);
// global address in a VGPR pair, GV mode (saddr = off).
__device__ __forceinline__ void async_ld16(void* lds, const void* gaddr) {
  unsigned lo = (unsigned)(unsigned long long)lds;
  asm volatile("global_load_async_to_lds_b128 %0, %1, off" ::"v"(lo),
               "v"(gaddr)
               : "memory");
}

#if __has_builtin(__builtin_amdgcn_s_wait_asynccnt)
#define WAIT_ASYNC(n) __builtin_amdgcn_s_wait_asynccnt(n)
#else
#define WAIT_ASYNC(n) asm volatile("s_wait_asynccnt %0" ::"i"(n))
#endif

// A fragment (16x32 bf16, ISA 7.12.2): lane&15 = row, lane>>4 selects K-half.
__device__ __forceinline__ bf16x16 load_a_frag(const unsigned short* rowp,
                                               int lane) {
  const int b0 = ((lane >> 4) & 1) * 8;
  u32x8 u;
#pragma unroll
  for (int i = 0; i < 4; ++i)
    u[i] = *(const unsigned int*)(rowp + b0 + 2 * i);
#pragma unroll
  for (int i = 0; i < 4; ++i)
    u[4 + i] = *(const unsigned int*)(rowp + b0 + 16 + 2 * i);
  return __builtin_bit_cast(bf16x16, u);
}

// B fragment (32x16 bf16): lane&15 = col; K = kh*8 + 2v + e (v<4), 16 + kh*8 +
// 2(v-4) + e (v>=4); colp is the K-contiguous column (weights stored (N,K)).
__device__ __forceinline__ bf16x16 load_b_frag(const unsigned short* colp,
                                               int lane) {
  const int b0 = ((lane >> 4) & 1) * 8;
  u32x8 u;
#pragma unroll
  for (int v = 0; v < 4; ++v)
    u[v] = *(const unsigned int*)(colp + b0 + 2 * v);
#pragma unroll
  for (int v = 0; v < 4; ++v)
    u[4 + v] = *(const unsigned int*)(colp + b0 + 16 + 2 * v);
  return __builtin_bit_cast(bf16x16, u);
}

// ---------------------------------------------------------------------------
// Fused GEMM: out[m] = relu(LN(X[m] @ W[m] + b[m])) (+skip) (*mask)
// X: (nm?,B,K) bf16, WT: (nm,N,K) bf16 (pre-transposed), out: (nm,B,N) bf16.
// 256 threads = 8 waves; each wave owns 16 complete rows. K streamed in 32-
// wide chunks through double-buffered LDS filled by async-to-LDS b128 loads.
// ---------------------------------------------------------------------------
template <int K, int N, bool ADD_SKIP, bool MASK_OUT>
__global__ __launch_bounds__(256) void gemm_ln_kernel(
    const unsigned short* __restrict__ X, size_t x_mstride,
    const unsigned short* __restrict__ WT, const float* __restrict__ bias,
    const float* __restrict__ gamma, const float* __restrict__ beta,
    const unsigned short* __restrict__ skipbuf,
    const int* __restrict__ domain, unsigned short* __restrict__ out) {
  constexpr int KC = 32, PK = 40;  // 80B pitch: aligned 16B segs, spread banks
  constexpr int NT = N / 16;
  constexpr int CNT = 2 + N / 64;  // async issues per wave per chunk
  __shared__ unsigned short sx[2][128 * PK];
  __shared__ unsigned short swt[2][N * PK];
  const int tid = threadIdx.x, lane = tid & 31, wv = tid >> 5;
  const int m = blockIdx.y;
  const int row0 = blockIdx.x * 128;
  const unsigned short* Xm = X + (size_t)m * x_mstride;
  const unsigned short* WmT = WT + (size_t)m * N * K;

  f32x8 acc[NT];
#pragma unroll
  for (int j = 0; j < NT; ++j)
#pragma unroll
    for (int r = 0; r < 8; ++r) acc[j][r] = 0.f;

  auto issue = [&](int bb, int kc) {
    for (int i = tid; i < 512; i += 256) {  // X: 128 rows x 4 segs
      int r = i >> 2, s2 = i & 3;
      async_ld16(&sx[bb][r * PK + s2 * 8],
                 Xm + (size_t)(row0 + r) * K + kc + s2 * 8);
    }
    for (int i = tid; i < N * 4; i += 256) {  // W: N cols x 4 segs
      int n = i >> 2, s2 = i & 3;
      async_ld16(&swt[bb][n * PK + s2 * 8], WmT + (size_t)n * K + kc + s2 * 8);
    }
  };

  issue(0, 0);
  int it = 0;
  for (int kc = 0; kc < K; kc += KC, ++it) {
    const int cur = it & 1;
    if (kc + KC < K) {
      issue(cur ^ 1, kc + KC);
      WAIT_ASYNC(CNT);  // next chunk in flight; current chunk complete
    } else {
      WAIT_ASYNC(0);
    }
    __syncthreads();
    const unsigned short* arow = &sx[cur][(wv * 16 + (lane & 15)) * PK];
    bf16x16 af = load_a_frag(arow, lane);
    bf16x16 bc = load_b_frag(&swt[cur][(lane & 15) * PK], lane);
#pragma unroll
    for (int j = 0; j < NT; ++j) {
      bf16x16 bn = bc;
      if (j + 1 < NT)
        bn = load_b_frag(&swt[cur][((j + 1) * 16 + (lane & 15)) * PK], lane);
      acc[j] = wmma_bf16(af, bc, acc[j]);
      bc = bn;
    }
    __syncthreads();
  }

  const int kh = lane >> 4, cl = lane & 15;
#pragma unroll
  for (int j = 0; j < NT; ++j) {
    float bb = bias[(size_t)m * N + j * 16 + cl];
#pragma unroll
    for (int r = 0; r < 8; ++r) acc[j][r] += bb;
  }
  // LayerNorm over N: xor-reduce across the 16 column lanes of each half.
  float s[8], q[8];
#pragma unroll
  for (int r = 0; r < 8; ++r) { s[r] = 0.f; q[r] = 0.f; }
#pragma unroll
  for (int j = 0; j < NT; ++j)
#pragma unroll
    for (int r = 0; r < 8; ++r) {
      float v = acc[j][r];
      s[r] += v;
      q[r] += v * v;
    }
#pragma unroll
  for (int msk = 1; msk <= 8; msk <<= 1)
#pragma unroll
    for (int r = 0; r < 8; ++r) {
      s[r] += __shfl_xor(s[r], msk);
      q[r] += __shfl_xor(q[r], msk);
    }
  float mean[8], rst[8];
#pragma unroll
  for (int r = 0; r < 8; ++r) {
    mean[r] = s[r] * (1.f / N);
    float var = q[r] * (1.f / N) - mean[r] * mean[r];
    rst[r] = rsqrtf(var + EPSF);
  }
#pragma unroll
  for (int j = 0; j < NT; ++j) {
    int col = j * 16 + cl;
    float gm = gamma[(size_t)m * N + col];
    float bt = beta[(size_t)m * N + col];
#pragma unroll
    for (int r = 0; r < 8; ++r) {
      int rowg = row0 + wv * 16 + r + kh * 8;
      float v = (acc[j][r] - mean[r]) * rst[r] * gm + bt;
      v = fmaxf(v, 0.f);
      if constexpr (ADD_SKIP) v += bf2f(skipbuf[(size_t)rowg * N + col]);
      if constexpr (MASK_OUT) v = (domain[rowg] == m) ? v : 0.f;
      out[((size_t)m * BN + rowg) * N + col] = f2bf(v);
    }
  }
}

// ---------------------------------------------------------------------------
// Slot: emb0[b] = X[b] @ (slot_W[d]*shared_W) + b_comb[d], d = domain[b].
// All 3 domains computed, predicated store. WcT: (3, 512, 256) transposed.
// ---------------------------------------------------------------------------
__global__ __launch_bounds__(256) void slot_kernel(
    const unsigned short* __restrict__ X, const unsigned short* __restrict__ WcT,
    const float* __restrict__ combB, const int* __restrict__ domain,
    unsigned short* __restrict__ emb0) {
  constexpr int K = 256, NTOT = 512, NB = 128, KC = 32, PK = 40, NT = NB / 16;
  constexpr int CNT = 2 + NB / 64;
  __shared__ unsigned short sx[2][128 * PK];
  __shared__ unsigned short swt[2][NB * PK];
  const int tid = threadIdx.x, lane = tid & 31, wv = tid >> 5;
  const int row0 = blockIdx.x * 128;
  const int ncol0 = blockIdx.y * NB;
  const int kh = lane >> 4, cl = lane & 15;

  for (int d = 0; d < 3; ++d) {
    const unsigned short* WmT = WcT + (size_t)d * NTOT * K;
    auto issue = [&](int bb, int kc) {
      for (int i = tid; i < 512; i += 256) {
        int r = i >> 2, s2 = i & 3;
        async_ld16(&sx[bb][r * PK + s2 * 8],
                   X + (size_t)(row0 + r) * K + kc + s2 * 8);
      }
      for (int i = tid; i < NB * 4; i += 256) {
        int n = i >> 2, s2 = i & 3;
        async_ld16(&swt[bb][n * PK + s2 * 8],
                   WmT + (size_t)(ncol0 + n) * K + kc + s2 * 8);
      }
    };
    f32x8 acc[NT];
#pragma unroll
    for (int j = 0; j < NT; ++j)
#pragma unroll
      for (int r = 0; r < 8; ++r) acc[j][r] = 0.f;

    issue(0, 0);
    int it = 0;
    for (int kc = 0; kc < K; kc += KC, ++it) {
      const int cur = it & 1;
      if (kc + KC < K) {
        issue(cur ^ 1, kc + KC);
        WAIT_ASYNC(CNT);
      } else {
        WAIT_ASYNC(0);
      }
      __syncthreads();
      const unsigned short* arow = &sx[cur][(wv * 16 + cl) * PK];
      bf16x16 af = load_a_frag(arow, lane);
      bf16x16 bc = load_b_frag(&swt[cur][cl * PK], lane);
#pragma unroll
      for (int j = 0; j < NT; ++j) {
        bf16x16 bn = bc;
        if (j + 1 < NT) bn = load_b_frag(&swt[cur][((j + 1) * 16 + cl) * PK], lane);
        acc[j] = wmma_bf16(af, bc, acc[j]);
        bc = bn;
      }
      __syncthreads();
    }
#pragma unroll
    for (int j = 0; j < NT; ++j) {
      int col = ncol0 + j * 16 + cl;
      float bb = combB[d * NTOT + col];
#pragma unroll
      for (int r = 0; r < 8; ++r) {
        int rowg = row0 + wv * 16 + r + kh * 8;
        if (domain[rowg] == d)
          emb0[(size_t)rowg * NTOT + col] = f2bf(acc[j][r] + bb);
      }
    }
  }
}

// ---------------------------------------------------------------------------
// Gate: logits = emb @ Wg + gate_b (N=48 = 6 groups of 8), softmax per 8.
// WgT: (48, 256) transposed. Small: synchronous staging.
// ---------------------------------------------------------------------------
__global__ __launch_bounds__(256) void gate_kernel(
    const unsigned short* __restrict__ X, const unsigned short* __restrict__ WgT,
    const float* __restrict__ gate_b, float* __restrict__ gate_out) {
  constexpr int K = 256, N = 48, KC = 64, NT = 3;
  __shared__ unsigned short sx[128 * KC];
  __shared__ unsigned short swt[N * KC];
  const int tid = threadIdx.x, lane = tid & 31, wv = tid >> 5;
  const int row0 = blockIdx.x * 128;
  const int kh = lane >> 4, cl = lane & 15;

  f32x8 acc[NT];
#pragma unroll
  for (int j = 0; j < NT; ++j)
#pragma unroll
    for (int r = 0; r < 8; ++r) acc[j][r] = 0.f;

  for (int kc = 0; kc < K; kc += KC) {
    unsigned int* dx = (unsigned int*)sx;
    for (int i = tid; i < 128 * (KC / 2); i += 256) {
      int r = i >> 5, c = i & 31;
      dx[i] = *(const unsigned int*)(X + (size_t)(row0 + r) * K + kc + 2 * c);
    }
    unsigned int* dw = (unsigned int*)swt;
    for (int i = tid; i < N * (KC / 2); i += 256) {
      int n = i >> 5, kp = i & 31;
      dw[n * (KC / 2) + kp] =
          *(const unsigned int*)(WgT + (size_t)n * K + kc + 2 * kp);
    }
    __syncthreads();
    const unsigned short* arow = sx + (size_t)(wv * 16 + cl) * KC;
#pragma unroll
    for (int kk = 0; kk < KC; kk += 32) {
      bf16x16 af = load_a_frag(arow + kk, lane);
#pragma unroll
      for (int j = 0; j < NT; ++j) {
        bf16x16 bfr = load_b_frag(swt + (size_t)(j * 16 + cl) * KC + kk, lane);
        acc[j] = wmma_bf16(af, bfr, acc[j]);
      }
    }
    __syncthreads();
  }
#pragma unroll
  for (int j = 0; j < NT; ++j) {
    int col = j * 16 + cl;
    float bb = gate_b[col];
#pragma unroll
    for (int r = 0; r < 8; ++r) {
      float v = acc[j][r] + bb;
      float mx = v;
#pragma unroll
      for (int msk = 1; msk <= 4; msk <<= 1)
        mx = fmaxf(mx, __shfl_xor(mx, msk));
      float ex = __expf(v - mx);
      float sm = ex;
#pragma unroll
      for (int msk = 1; msk <= 4; msk <<= 1) sm += __shfl_xor(sm, msk);
      int rowg = row0 + wv * 16 + r + kh * 8;
      gate_out[(size_t)rowg * 48 + col] = ex / sm;
    }
  }
}

// ---------------------------------------------------------------------------
// Mix + tower layer 1: task_fea = gate·fea + dom[g%3]; h = task_fea @ tW1 + b.
// Writes h (f32) and deterministic per-block col sums / sumsq partials.
// tw1T: (6, 64, 64) transposed (n,k).
// ---------------------------------------------------------------------------
__global__ __launch_bounds__(256) void mix_tw1_kernel(
    const float* __restrict__ gate_f, const unsigned short* __restrict__ fea,
    const unsigned short* __restrict__ domv,
    const unsigned short* __restrict__ tw1T, const float* __restrict__ tb1,
    float* __restrict__ hpre, float* __restrict__ gps,
    float* __restrict__ gpq) {
  __shared__ unsigned short T[128 * 64];
  __shared__ unsigned short Wt[64 * 64];
  __shared__ float redS[8 * 64];
  __shared__ float redQ[8 * 64];
  const int tid = threadIdx.x, lane = tid & 31, wv = tid >> 5;
  const int g = blockIdx.y, d = g % 3;
  const int row0 = blockIdx.x * 128;

  {  // task_fea tile (VALU; 8 MACs per element)
    int row = tid >> 1;
    int o0 = (tid & 1) * 32;
    int rowg = row0 + row;
    float gv[8];
    const float* gp = gate_f + (size_t)rowg * 48 + g * 8;
#pragma unroll
    for (int e = 0; e < 8; ++e) gv[e] = gp[e];
    const unsigned short* dp = domv + ((size_t)d * BN + rowg) * 64;
    for (int o = o0; o < o0 + 32; ++o) {
      float a = bf2f(dp[o]);
#pragma unroll
      for (int e = 0; e < 8; ++e)
        a += gv[e] * bf2f(fea[((size_t)e * BN + rowg) * 64 + o]);
      T[row * 64 + o] = f2bf(a);
    }
  }
  {
    unsigned int* dw = (unsigned int*)Wt;
    const unsigned int* sw = (const unsigned int*)(tw1T + (size_t)g * 4096);
    for (int i = tid; i < 2048; i += 256) dw[i] = sw[i];
  }
  __syncthreads();

  f32x8 acc[4];
#pragma unroll
  for (int j = 0; j < 4; ++j)
#pragma unroll
    for (int r = 0; r < 8; ++r) acc[j][r] = 0.f;
  const int cl = lane & 15, kh = lane >> 4;
  const unsigned short* arow = T + (size_t)(wv * 16 + cl) * 64;
#pragma unroll
  for (int kk = 0; kk < 64; kk += 32) {
    bf16x16 af = load_a_frag(arow + kk, lane);
#pragma unroll
    for (int j = 0; j < 4; ++j) {
      bf16x16 bfr = load_b_frag(Wt + (size_t)(j * 16 + cl) * 64 + kk, lane);
      acc[j] = wmma_bf16(af, bfr, acc[j]);
    }
  }

#pragma unroll
  for (int j = 0; j < 4; ++j) {
    int col = j * 16 + cl;
    float bb = tb1[g * 64 + col];
    float s = 0.f, q = 0.f;
#pragma unroll
    for (int r = 0; r < 8; ++r) {
      int rowg = row0 + wv * 16 + r + kh * 8;
      float v = acc[j][r] + bb;
      hpre[((size_t)g * BN + rowg) * 64 + col] = v;
      s += v;
      q += v * v;
    }
    s += __shfl_xor(s, 16);
    q += __shfl_xor(q, 16);
    if (lane < 16) {
      redS[wv * 64 + col] = s;
      redQ[wv * 64 + col] = q;
    }
  }
  __syncthreads();
  if (tid < 64) {
    float s = 0.f, q = 0.f;
#pragma unroll
    for (int w = 0; w < 8; ++w) {
      s += redS[w * 64 + tid];
      q += redQ[w * 64 + tid];
    }
    gps[((size_t)g * gridDim.x + blockIdx.x) * 64 + tid] = s;
    gpq[((size_t)g * gridDim.x + blockIdx.x) * 64 + tid] = q;
  }
}

// ---------------------------------------------------------------------------
// Small kernels: prep / gather / stats / output
// ---------------------------------------------------------------------------
// Cast f32 (nm,K,N) -> bf16 transposed (nm,N,K).
__global__ void castT_kernel(const float* __restrict__ in,
                             unsigned short* __restrict__ out, int nm, int K,
                             int N) {
  int i = blockIdx.x * 256 + threadIdx.x;
  if (i < nm * K * N) {
    int m = i / (K * N), r = i % (K * N), k = r / N, n = r % N;
    out[((size_t)m * N + n) * K + k] = f2bf(in[i]);
  }
}

__global__ void combineT_kernel(const float* __restrict__ slotW,
                                const float* __restrict__ sharedW,
                                unsigned short* __restrict__ out) {
  int i = blockIdx.x * 256 + threadIdx.x;
  if (i < 3 * 256 * 512) {
    int d = i / (256 * 512), r = i % (256 * 512), k = r / 512, n = r % 512;
    out[((size_t)d * 512 + n) * 256 + k] = f2bf(slotW[i] * sharedW[r]);
  }
}

__global__ void comb_bias_kernel(const float* __restrict__ slot_b,
                                 const float* __restrict__ shared_b,
                                 float* __restrict__ combB) {
  int i = blockIdx.x * 256 + threadIdx.x;
  if (i < 3 * 512) combB[i] = slot_b[i] + shared_b[i & 511];
}

// gate_W (6,256,8) -> bf16 (48,256): row n = g*8+e holds K-contiguous slice.
__global__ void gatepackT_kernel(const float* __restrict__ gw,
                                 unsigned short* __restrict__ out) {
  int i = blockIdx.x * 256 + threadIdx.x;
  if (i < 6 * 256 * 8) {
    int g = i >> 11, r = i & 2047, k = r >> 3, e = r & 7;
    out[((size_t)(g * 8 + e)) * 256 + k] = f2bf(gw[i]);
  }
}

__global__ void embed_kernel(const int* __restrict__ x,
                             const float* __restrict__ tables,
                             unsigned short* __restrict__ X) {
  int idx = blockIdx.x * 256 + threadIdx.x;  // over B*16
  int b = idx >> 4, f = idx & 15;
  int id = x[(size_t)f * BN + b];
  const float* src = tables + ((size_t)f * 10000 + id) * 16;
  unsigned int* dst = (unsigned int*)(X + (size_t)b * 256 + f * 16);
#pragma unroll
  for (int w = 0; w < 8; ++w)
    dst[w] = (unsigned int)f2bf(src[2 * w]) |
             ((unsigned int)f2bf(src[2 * w + 1]) << 16);
}

__global__ void stats_kernel(const float* __restrict__ gps,
                             const float* __restrict__ gpq, int nblk,
                             float* __restrict__ mu, float* __restrict__ rstd) {
  int i = threadIdx.x;  // 384 = 6*64
  if (i < 384) {
    int g = i >> 6, col = i & 63;
    float s = 0.f, q = 0.f;
    for (int b = 0; b < nblk; ++b) {
      s += gps[((size_t)g * nblk + b) * 64 + col];
      q += gpq[((size_t)g * nblk + b) * 64 + col];
    }
    float m = s * (1.f / BN);
    float v = q * (1.f / BN) - m * m;
    mu[i] = m;
    rstd[i] = rsqrtf(v + EPSF);
  }
}

__global__ void out_kernel(const float* __restrict__ hpre,
                           const float* __restrict__ mu,
                           const float* __restrict__ rstd,
                           const float* __restrict__ tg,
                           const float* __restrict__ tbeta,
                           const float* __restrict__ tW2,
                           const float* __restrict__ tb2,
                           const int* __restrict__ domain,
                           float* __restrict__ out) {
  int b = blockIdx.x * 256 + threadIdx.x;
  float res[6];
#pragma unroll
  for (int g = 0; g < 6; ++g) {
    const float* hp = hpre + ((size_t)g * BN + b) * 64;
    float a = 0.f;
#pragma unroll 16
    for (int o = 0; o < 64; ++o) {
      float hn = (hp[o] - mu[g * 64 + o]) * rstd[g * 64 + o] * tg[g * 64 + o] +
                 tbeta[g * 64 + o];
      hn = fmaxf(hn, 0.f);
      a += hn * tW2[g * 64 + o];
    }
    res[g] = 1.f / (1.f + __expf(-(a + tb2[g])));
  }
  int d = domain[b];
  out[(size_t)b * 2 + 0] = res[d];
  out[(size_t)b * 2 + 1] = res[3 + d];
}

// ---------------------------------------------------------------------------
extern "C" void kernel_launch(void* const* d_in, const int* in_sizes, int n_in,
                              void* d_out, int out_size, void* d_ws,
                              size_t ws_size, hipStream_t stream) {
  (void)in_sizes; (void)n_in; (void)out_size; (void)ws_size;
  const int* x = (const int*)d_in[0];
  const float* emb_tables = (const float*)d_in[1];
  const float* skip_W = (const float*)d_in[2];
  const float* skip_b = (const float*)d_in[3];
  const float* skip_g = (const float*)d_in[4];
  const float* skip_be = (const float*)d_in[5];
  const float* shared_W = (const float*)d_in[6];
  const float* shared_b = (const float*)d_in[7];
  const float* slot_W = (const float*)d_in[8];
  const float* slot_b = (const float*)d_in[9];
  const float* star_W = (const float*)d_in[10];
  const float* star_b = (const float*)d_in[11];
  const float* star_g = (const float*)d_in[12];
  const float* star_be = (const float*)d_in[13];
  const float* gate_W = (const float*)d_in[14];
  const float* gate_b = (const float*)d_in[15];
  const float* eW1 = (const float*)d_in[16];
  const float* eb1 = (const float*)d_in[17];
  const float* eg1 = (const float*)d_in[18];
  const float* ebe1 = (const float*)d_in[19];
  const float* eW2 = (const float*)d_in[20];
  const float* eb2 = (const float*)d_in[21];
  const float* eg2 = (const float*)d_in[22];
  const float* ebe2 = (const float*)d_in[23];
  const float* dW1 = (const float*)d_in[24];
  const float* db1 = (const float*)d_in[25];
  const float* dg1 = (const float*)d_in[26];
  const float* dbe1 = (const float*)d_in[27];
  const float* dW2 = (const float*)d_in[28];
  const float* db2 = (const float*)d_in[29];
  const float* dg2 = (const float*)d_in[30];
  const float* dbe2 = (const float*)d_in[31];
  const float* tW1 = (const float*)d_in[32];
  const float* tb1 = (const float*)d_in[33];
  const float* tg = (const float*)d_in[34];
  const float* tbeta = (const float*)d_in[35];
  const float* tW2 = (const float*)d_in[36];
  const float* tb2 = (const float*)d_in[37];

  char* w = (char*)d_ws;
  size_t off = 0;
  auto A = [&](size_t bytes) -> char* {
    char* p = w + off;
    off = (off + bytes + 255) & ~(size_t)255;
    return p;
  };
  unsigned short* Xbf = (unsigned short*)A((size_t)BN * 256 * 2);
  unsigned short* skipbf = (unsigned short*)A((size_t)BN * 256 * 2);
  unsigned short* emb0bf = (unsigned short*)A((size_t)BN * 512 * 2);
  unsigned short* embbf = (unsigned short*)A((size_t)BN * 256 * 2);
  float* gatef = (float*)A((size_t)BN * 48 * 4);
  unsigned short* h1e = (unsigned short*)A((size_t)8 * BN * 128 * 2);
  unsigned short* feab = (unsigned short*)A((size_t)8 * BN * 64 * 2);
  unsigned short* h1d = (unsigned short*)A((size_t)3 * BN * 128 * 2);
  unsigned short* domb = (unsigned short*)A((size_t)3 * BN * 64 * 2);
  float* hpre = (float*)A((size_t)6 * BN * 64 * 4);
  unsigned short* wskipT = (unsigned short*)A(256 * 256 * 2);
  unsigned short* wcombT = (unsigned short*)A(3 * 256 * 512 * 2);
  float* combB = (float*)A(3 * 512 * 4);
  unsigned short* wstarT = (unsigned short*)A(512 * 256 * 2);
  unsigned short* wgateT = (unsigned short*)A(48 * 256 * 2);
  unsigned short* we1T = (unsigned short*)A(8 * 256 * 128 * 2);
  unsigned short* we2T = (unsigned short*)A(8 * 128 * 64 * 2);
  unsigned short* wd1T = (unsigned short*)A(3 * 256 * 128 * 2);
  unsigned short* wd2T = (unsigned short*)A(3 * 128 * 64 * 2);
  unsigned short* wt1T = (unsigned short*)A(6 * 64 * 64 * 2);
  float* gps = (float*)A((size_t)6 * 512 * 64 * 4);
  float* gpq = (float*)A((size_t)6 * 512 * 64 * 4);
  float* mu = (float*)A(384 * 4);
  float* rstd = (float*)A(384 * 4);

  const int* domain = x + (size_t)15 * BN;
  auto nblks = [](int n) { return (n + 255) / 256; };

  // weight prep: cast to bf16, transpose to (N,K)
  castT_kernel<<<nblks(256 * 256), 256, 0, stream>>>(skip_W, wskipT, 1, 256, 256);
  combineT_kernel<<<nblks(3 * 256 * 512), 256, 0, stream>>>(slot_W, shared_W, wcombT);
  comb_bias_kernel<<<nblks(3 * 512), 256, 0, stream>>>(slot_b, shared_b, combB);
  castT_kernel<<<nblks(512 * 256), 256, 0, stream>>>(star_W, wstarT, 1, 512, 256);
  gatepackT_kernel<<<nblks(6 * 256 * 8), 256, 0, stream>>>(gate_W, wgateT);
  castT_kernel<<<nblks(8 * 256 * 128), 256, 0, stream>>>(eW1, we1T, 8, 256, 128);
  castT_kernel<<<nblks(8 * 128 * 64), 256, 0, stream>>>(eW2, we2T, 8, 128, 64);
  castT_kernel<<<nblks(3 * 256 * 128), 256, 0, stream>>>(dW1, wd1T, 3, 256, 128);
  castT_kernel<<<nblks(3 * 128 * 64), 256, 0, stream>>>(dW2, wd2T, 3, 128, 64);
  castT_kernel<<<nblks(6 * 64 * 64), 256, 0, stream>>>(tW1, wt1T, 6, 64, 64);

  // embedding gather -> X (B,256) bf16
  embed_kernel<<<(BN * 16) / 256, 256, 0, stream>>>(x, emb_tables, Xbf);

  const int MB = BN / 128;  // 512 row-tiles
  // skip = relu(LN(X @ skip_W + b))
  gemm_ln_kernel<256, 256, false, false><<<dim3(MB, 1), 256, 0, stream>>>(
      Xbf, 0, wskipT, skip_b, skip_g, skip_be, nullptr, nullptr, skipbf);
  // emb0 = X @ (slot*shared)[domain] + b_comb[domain]
  slot_kernel<<<dim3(MB, 4), 256, 0, stream>>>(Xbf, wcombT, combB, domain, emb0bf);
  // emb = relu(LN(emb0 @ star_W + b)) + skip
  gemm_ln_kernel<512, 256, true, false><<<dim3(MB, 1), 256, 0, stream>>>(
      emb0bf, 0, wstarT, star_b, star_g, star_be, skipbf, nullptr, embbf);
  // gate = softmax(emb @ gate_W + b) per 8
  gate_kernel<<<MB, 256, 0, stream>>>(embbf, wgateT, gate_b, gatef);
  // expert MLP
  gemm_ln_kernel<256, 128, false, false><<<dim3(MB, 8), 256, 0, stream>>>(
      embbf, 0, we1T, eb1, eg1, ebe1, nullptr, nullptr, h1e);
  gemm_ln_kernel<128, 64, false, false><<<dim3(MB, 8), 256, 0, stream>>>(
      h1e, (size_t)BN * 128, we2T, eb2, eg2, ebe2, nullptr, nullptr, feab);
  // domain MLP (masked output)
  gemm_ln_kernel<256, 128, false, false><<<dim3(MB, 3), 256, 0, stream>>>(
      embbf, 0, wd1T, db1, dg1, dbe1, nullptr, nullptr, h1d);
  gemm_ln_kernel<128, 64, false, true><<<dim3(MB, 3), 256, 0, stream>>>(
      h1d, (size_t)BN * 128, wd2T, db2, dg2, dbe2, nullptr, domain, domb);
  // mix + tower layer1 + batch-stat partials
  mix_tw1_kernel<<<dim3(MB, 6), 256, 0, stream>>>(gatef, feab, domb, wt1T, tb1,
                                                  hpre, gps, gpq);
  stats_kernel<<<1, 384, 0, stream>>>(gps, gpq, MB, mu, rstd);
  out_kernel<<<BN / 256, 256, 0, stream>>>(hpre, mu, rstd, tg, tbeta, tW2, tb2,
                                           domain, (float*)d_out);
}
